// CausalSelfAttention_80805514707513
// MI455X (gfx1250) — matrix operand order repaired and checked by reading, not compiled, not run on hardware
//
#include <hip/hip_runtime.h>

// ---------------------------------------------------------------------------
// Problem constants (from reference): B=2, T=2048, C=1536, H=12, KV=4, D=128
// ---------------------------------------------------------------------------
static constexpr int Bb  = 2;
static constexpr int Tt  = 2048;
static constexpr int Cc  = 1536;
static constexpr int Hh  = 12;
static constexpr int KVh = 4;
static constexpr int Dd  = 128;
static constexpr int GCH = 12;           // VE gate channels
static constexpr int BT  = Bb * Tt;      // 4096 rows
static constexpr int NQKV = Hh * Dd + 2 * KVh * Dd; // 2560

typedef __attribute__((ext_vector_type(16))) __bf16 v16bf;
typedef __attribute__((ext_vector_type(8)))  __bf16 v8bf;
typedef __attribute__((ext_vector_type(8)))  float  v8f;

// ---------------------------------------------------------------------------
// Fragment loaders following CDNA5 ISA 7.12.2 wave32 layouts (bf16, 16x16x32)
// A (16x32, MxK): lane&15 = M; hi=lane>>4; j<8 -> K=hi*8+j, j>=8 -> K=16+hi*8+(j-8)
// B (32x16, KxN): lane&15 = N; j<8 -> K=hi*8+j, j>=8 -> K=16+hi*8+(j-8)
// C/D (16x16 f32): lane&15 = N; acc[r] -> M = hi*8 + r
// ---------------------------------------------------------------------------
__device__ __forceinline__ v16bf load_frag_a(const __bf16* base, int ld, int lane) {
  const int m = lane & 15, hi = lane >> 4;
  const __bf16* p = base + m * ld + hi * 8;
  v8bf lo = *(const v8bf*)(p);
  v8bf hh = *(const v8bf*)(p + 16);
  v16bf r;
#pragma unroll
  for (int j = 0; j < 8; ++j) { r[j] = lo[j]; r[8 + j] = hh[j]; }
  return r;
}

__device__ __forceinline__ v16bf load_frag_b(const __bf16* base, int ld, int lane) {
  const int n = lane & 15, hi = lane >> 4;
  const __bf16* p = base + n * ld + hi * 8;
  v8bf b0 = *(const v8bf*)(p);
  v8bf b1 = *(const v8bf*)(p + 16);
  v16bf r;
#pragma unroll
  for (int j = 0; j < 8; ++j) { r[j] = b0[j]; r[8 + j] = b1[j]; }
  return r;
}

// ---------------------------------------------------------------------------
// f32 -> bf16 conversion (grid-stride)
// ---------------------------------------------------------------------------
__global__ void cvt_f32_bf16(const float* __restrict__ s, __bf16* __restrict__ d, int n) {
  int i = blockIdx.x * blockDim.x + threadIdx.x;
  int stride = gridDim.x * blockDim.x;
  for (; i < n; i += stride) d[i] = (__bf16)s[i];
}

// ---------------------------------------------------------------------------
// Tiled bf16 GEMM:  C[M,N] = A[M,K] * W[N,K]^T   (f32 accumulate via WMMA)
// Block = 256 threads (8 waves). Tile 128x128, K-chunks of 64.
// Register-staged double buffering: next global tile is loaded into VGPRs
// while WMMAs run on the current LDS tile. B-fragments batched in groups of 4
// so DS loads pipeline ahead of back-to-back WMMAs.
// Assumes M%128==0, N%128==0, K%64==0 (true for all uses here).
// ---------------------------------------------------------------------------
static constexpr int BK  = 64;
static constexpr int LDT = BK + 8;  // 72 halves; 144B stride (16B multiple)

__global__ __launch_bounds__(256)
void gemm_bf16_nt(const __bf16* __restrict__ A, const __bf16* __restrict__ W,
                  float* __restrict__ C, int M, int N, int K) {
  __shared__ __bf16 lA[128 * LDT];
  __shared__ __bf16 lB[128 * LDT];
  const int tid  = threadIdx.x;
  const int lane = tid & 31, wv = tid >> 5;
  const int m0 = blockIdx.y * 128, n0 = blockIdx.x * 128;

  v8f acc[8] = {};

  const int lr = tid >> 1;          // load row 0..127
  const int lc = (tid & 1) * 32;    // load col 0 / 32
  const __bf16* gA = A + (size_t)(m0 + lr) * K + lc;
  const __bf16* gB = W + (size_t)(n0 + lr) * K + lc;
  __bf16* sA = lA + lr * LDT + lc;
  __bf16* sB = lB + lr * LDT + lc;

  v8bf rA[4], rB[4];
#pragma unroll
  for (int j = 0; j < 4; ++j) {     // stage tile 0
    rA[j] = *(const v8bf*)(gA + j * 8);
    rB[j] = *(const v8bf*)(gB + j * 8);
  }

  for (int k0 = 0; k0 < K; k0 += BK) {
#pragma unroll
    for (int j = 0; j < 4; ++j) {   // commit staged tile to LDS
      *(v8bf*)(sA + j * 8) = rA[j];
      *(v8bf*)(sB + j * 8) = rB[j];
    }
    __syncthreads();

    if (k0 + BK < K) {              // stage next tile while WMMAs run
#pragma unroll
      for (int j = 0; j < 4; ++j) {
        rA[j] = *(const v8bf*)(gA + k0 + BK + j * 8);
        rB[j] = *(const v8bf*)(gB + k0 + BK + j * 8);
      }
      if (k0 + 2 * BK < K) {        // gfx1250 global_prefetch_b8
        __builtin_prefetch(gA + k0 + 2 * BK, 0, 1);
        __builtin_prefetch(gB + k0 + 2 * BK, 0, 1);
      }
    }

#pragma unroll
    for (int kk = 0; kk < BK; kk += 32) {
      v16bf af = load_frag_a(lA + (wv * 16) * LDT + kk, LDT, lane);
#pragma unroll
      for (int sh = 0; sh < 2; ++sh) {
        v16bf bf4[4];
#pragma unroll
        for (int s = 0; s < 4; ++s)
          bf4[s] = load_frag_b(lB + ((sh * 4 + s) * 16) * LDT + kk, LDT, lane);
#pragma unroll
        for (int s = 0; s < 4; ++s)
          acc[sh * 4 + s] = __builtin_amdgcn_wmma_f32_16x16x32_bf16(
              false, af, false, bf4[s], (short)0, acc[sh * 4 + s], false, false);
      }
    }
    __syncthreads();
  }

  const int hi = lane >> 4, nn = lane & 15;
#pragma unroll
  for (int s = 0; s < 8; ++s)
#pragma unroll
    for (int r = 0; r < 8; ++r)
      C[(size_t)(m0 + wv * 16 + hi * 8 + r) * N + n0 + s * 16 + nn] = acc[s][r];
}

// ---------------------------------------------------------------------------
// Per-token epilogue: gate = 3*sigmoid(x[:12] @ Wg^T); v += gate*ve;
// q,k = rmsnorm(rope(q,k)) * 1.2 ; emit bf16 in [b][head][t][d] layout.
// One block (128 threads) per token.
// ---------------------------------------------------------------------------
__global__ __launch_bounds__(128)
void qkv_post(const float* __restrict__ x, const float* __restrict__ ve,
              const float* __restrict__ cosb, const float* __restrict__ sinb,
              const float* __restrict__ Wg, const float* __restrict__ qkv,
              __bf16* __restrict__ qo, __bf16* __restrict__ ko,
              __bf16* __restrict__ vo) {
  const int bt = blockIdx.x;
  const int b = bt / Tt, t = bt % Tt;
  const int d = threadIdx.x;                 // 0..127
  const int lane = d & 31, wv = d >> 5;

  __shared__ float gate[KVh];
  __shared__ float red[4];

  const float* row = qkv + (size_t)bt * NQKV;

  if (d < KVh) {
    float g = 0.f;
    const float* xr = x + (size_t)bt * Cc;
#pragma unroll
    for (int c = 0; c < GCH; ++c) g += xr[c] * Wg[d * GCH + c];
    gate[d] = 3.f / (1.f + __expf(-g));
  }
  __syncthreads();

#pragma unroll
  for (int kv = 0; kv < KVh; ++kv) {
    float vv = row[Hh * Dd + KVh * Dd + kv * Dd + d] +
               gate[kv] * ve[((size_t)bt * KVh + kv) * Dd + d];
    vo[(((size_t)b * KVh + kv) * Tt + t) * Dd + d] = (__bf16)vv;
  }

  const float cs = cosb[t * 64 + (d & 63)];
  const float sn = sinb[t * 64 + (d & 63)];

  for (int h = 0; h < Hh + KVh; ++h) {
    const float* hb = (h < Hh) ? (row + h * Dd)
                               : (row + Hh * Dd + (h - Hh) * Dd);
    float r;
    if (d < 64) { float x1 = hb[d], x2 = hb[d + 64];   r =  x1 * cs + x2 * sn; }
    else        { float x1 = hb[d - 64], x2 = hb[d];   r = -x1 * sn + x2 * cs; }
    float ss = r * r;
#pragma unroll
    for (int o = 16; o > 0; o >>= 1) ss += __shfl_xor(ss, o, 32);
    if (lane == 0) red[wv] = ss;
    __syncthreads();
    float tot = red[0] + red[1] + red[2] + red[3];
    float out = r * rsqrtf(tot * (1.f / 128.f) + 1.1920929e-7f) * 1.2f;
    if (h < Hh) qo[(((size_t)b * Hh  + h)        * Tt + t) * Dd + d] = (__bf16)out;
    else        ko[(((size_t)b * KVh + (h - Hh)) * Tt + t) * Dd + d] = (__bf16)out;
    __syncthreads();
  }
}

// ---------------------------------------------------------------------------
// Flash attention (windowed causal, GQA rep=3), all-WMMA.
// Block = 256 threads (8 waves), 128 query rows per block, key blocks of 64.
// K tile is fetched with gfx1250 async-to-LDS DMA (ASYNCcnt path); V goes
// through VGPRs because it is transposed into [d][key] layout for the P*V
// B-fragments. Online softmax in f32; P round-trips through LDS to convert
// from the C/D layout into the A-fragment layout.
// ---------------------------------------------------------------------------
static constexpr int LDK = Dd + 8;  // 136
static constexpr int LDV = 64 + 8;  // 72
static constexpr int LDP = 64 + 8;  // 72

__global__ __launch_bounds__(256)
void flash_attn(const __bf16* __restrict__ Q, const __bf16* __restrict__ Kb,
                const __bf16* __restrict__ Vb, const int* __restrict__ winp,
                __bf16* __restrict__ Y) {
  const int b = blockIdx.z, h = blockIdx.y;
  const int qt0 = blockIdx.x * 128;
  const int kvh = h / (Hh / KVh);
  const int tid = threadIdx.x, lane = tid & 31, wv = tid >> 5;
  const int hi = lane >> 4, nn = lane & 15;
  const int win = winp[0];

  __shared__ __bf16 lK[64 * LDK];     // K block  [key][d]
  __shared__ __bf16 lVt[Dd * LDV];    // V block transposed [d][key]
  __shared__ __bf16 lP[128 * LDP];    // P tile   [qrow][key]

  // Q fragments: 4 d-chunks of 32, kept in registers for the whole block.
  v16bf qf[4];
  {
    const int tq = qt0 + wv * 16 + nn;
    const __bf16* qr = Q + (((size_t)b * Hh + h) * Tt + tq) * Dd;
#pragma unroll
    for (int c = 0; c < 4; ++c) {
      const __bf16* p = qr + c * 32 + hi * 8;
      v8bf lo = *(const v8bf*)p;
      v8bf hh = *(const v8bf*)(p + 16);
#pragma unroll
      for (int j = 0; j < 8; ++j) { qf[c][j] = lo[j]; qf[c][8 + j] = hh[j]; }
    }
  }

  float m8[8], l8[8];
  v8f oa[8] = {};
#pragma unroll
  for (int r = 0; r < 8; ++r) { m8[r] = -3.0e38f; l8[r] = 0.f; }

  const float scale = 0.08838834764831845f;   // 1/sqrt(128)

  int lo = qt0 - win; if (lo < 0) lo = 0;
  const int kb_lo = lo >> 6;
  const int kb_hi = (qt0 + 127) >> 6;

  for (int kb = kb_lo; kb <= kb_hi; ++kb) {
    const int s0 = kb * 64;
    // --- fetch K block via async DMA to LDS; V manually transposed ---------
    {
      const int r  = tid >> 2;              // 0..63 key row
      const int c0 = (tid & 3) * 32;        // 4 threads cover 128 d's
      const __bf16* gk = Kb + (((size_t)b * KVh + kvh) * Tt + s0 + r) * Dd + c0;
      const __bf16* gv = Vb + (((size_t)b * KVh + kvh) * Tt + s0 + r) * Dd + c0;
#pragma unroll
      for (int j = 0; j < 4; ++j) {
        unsigned lds_off = (unsigned)(size_t)(lK + r * LDK + c0 + j * 8);
        unsigned long long ga = (unsigned long long)(size_t)(gk + j * 8);
        asm volatile("global_load_async_to_lds_b128 %0, %1, off"
                     :: "v"(lds_off), "v"(ga) : "memory");
      }
#pragma unroll
      for (int j = 0; j < 4; ++j) {
        v8bf vvv = *(const v8bf*)(gv + j * 8);
#pragma unroll
        for (int e = 0; e < 8; ++e) lVt[(c0 + j * 8 + e) * LDV + r] = vvv[e];
      }
    }
    asm volatile("s_wait_asynccnt 0x0" ::: "memory");
    __syncthreads();

    // --- S = Q * K^T: batch 4 B-fragments per k-chunk, 4 independent WMMAs -
    v8f sa[4];
#pragma unroll
    for (int s = 0; s < 4; ++s) sa[s] = (v8f){};
#pragma unroll
    for (int c = 0; c < 4; ++c) {
      v16bf bf4[4];
#pragma unroll
      for (int s = 0; s < 4; ++s)
        bf4[s] = load_frag_b(lK + (s * 16) * LDK + c * 32, LDK, lane);
#pragma unroll
      for (int s = 0; s < 4; ++s)
        sa[s] = __builtin_amdgcn_wmma_f32_16x16x32_bf16(
            false, qf[c], false, bf4[s], (short)0, sa[s], false, false);
    }

    // --- masked online softmax ---------------------------------------------
    float mnew[8];
#pragma unroll
    for (int r = 0; r < 8; ++r) mnew[r] = m8[r];

#pragma unroll
    for (int s = 0; s < 4; ++s) {
#pragma unroll
      for (int r = 0; r < 8; ++r) {
        const int row = qt0 + wv * 16 + hi * 8 + r;
        const int col = s0 + s * 16 + nn;
        float v = sa[s][r] * scale;
        if (col > row || (row - col) > win) v = -3.0e38f;
        sa[s][r] = v;
        float mx = v;
#pragma unroll
        for (int o = 1; o < 16; o <<= 1) mx = fmaxf(mx, __shfl_xor(mx, o, 32));
        mnew[r] = fmaxf(mnew[r], mx);
      }
    }

    float alpha[8], rsum[8];
#pragma unroll
    for (int r = 0; r < 8; ++r) {
      alpha[r] = __expf(m8[r] - mnew[r]);
      m8[r] = mnew[r];
      rsum[r] = 0.f;
    }

#pragma unroll
    for (int s = 0; s < 4; ++s) {
#pragma unroll
      for (int r = 0; r < 8; ++r) {
        float sval = sa[s][r];
        float p = (sval < -1.0e37f) ? 0.f : __expf(sval - mnew[r]);
        lP[(wv * 16 + hi * 8 + r) * LDP + s * 16 + nn] = (__bf16)p;
        float ps = p;
#pragma unroll
        for (int o = 1; o < 16; o <<= 1) ps += __shfl_xor(ps, o, 32);
        rsum[r] += ps;
      }
    }
#pragma unroll
    for (int r = 0; r < 8; ++r) l8[r] = l8[r] * alpha[r] + rsum[r];
#pragma unroll
    for (int ds = 0; ds < 8; ++ds)
#pragma unroll
      for (int r = 0; r < 8; ++r) oa[ds][r] *= alpha[r];

    // --- O += P * V  (DS ops from same wave are in-order; lP rows private) --
#pragma unroll
    for (int c2 = 0; c2 < 2; ++c2) {
      v16bf pf = load_frag_a(lP + (wv * 16) * LDP + c2 * 32, LDP, lane);
#pragma unroll
      for (int half = 0; half < 2; ++half) {
        v16bf vf4[4];
#pragma unroll
        for (int s = 0; s < 4; ++s)
          vf4[s] = load_frag_b(lVt + ((half * 4 + s) * 16) * LDV + c2 * 32, LDV, lane);
#pragma unroll
        for (int s = 0; s < 4; ++s)
          oa[half * 4 + s] = __builtin_amdgcn_wmma_f32_16x16x32_bf16(
              false, pf, false, vf4[s], (short)0, oa[half * 4 + s], false, false);
      }
    }
    __syncthreads();
  }

  float inv[8];
#pragma unroll
  for (int r = 0; r < 8; ++r) inv[r] = 1.f / l8[r];
#pragma unroll
  for (int ds = 0; ds < 8; ++ds)
#pragma unroll
    for (int r = 0; r < 8; ++r) {
      const int trow = qt0 + wv * 16 + hi * 8 + r;
      Y[(((size_t)b * Tt + trow) * Hh + h) * Dd + ds * 16 + nn] =
          (__bf16)(oa[ds][r] * inv[r]);
    }
}

// ---------------------------------------------------------------------------
// Host side
// ---------------------------------------------------------------------------
extern "C" void kernel_launch(void* const* d_in, const int* in_sizes, int n_in,
                              void* d_out, int out_size, void* d_ws, size_t ws_size,
                              hipStream_t stream) {
  (void)in_sizes; (void)n_in; (void)out_size; (void)ws_size;

  const float* x     = (const float*)d_in[0];
  const float* ve    = (const float*)d_in[1];
  const float* cosb  = (const float*)d_in[2];
  const float* sinb  = (const float*)d_in[3];
  const float* Wq    = (const float*)d_in[4];
  const float* Wk    = (const float*)d_in[5];
  const float* Wv    = (const float*)d_in[6];
  const float* Wproj = (const float*)d_in[7];
  const float* Wg    = (const float*)d_in[8];
  const int*   win   = (const int*)d_in[9];
  float* out = (float*)d_out;

  char* ws = (char*)d_ws;
  size_t off = 0;
  auto alloc = [&](size_t bytes) -> void* {
    void* p = ws + off;
    off = (off + bytes + 255) & ~(size_t)255;
    return p;
  };

  __bf16* x_bf     = (__bf16*)alloc((size_t)BT * Cc * 2);
  __bf16* wqkv_bf  = (__bf16*)alloc((size_t)NQKV * Cc * 2);
  __bf16* wproj_bf = (__bf16*)alloc((size_t)Cc * Cc * 2);
  float*  qkv      = (float*) alloc((size_t)BT * NQKV * 4);
  __bf16* q_bf     = (__bf16*)alloc((size_t)Bb * Hh  * Tt * Dd * 2);
  __bf16* k_bf     = (__bf16*)alloc((size_t)Bb * KVh * Tt * Dd * 2);
  __bf16* v_bf     = (__bf16*)alloc((size_t)Bb * KVh * Tt * Dd * 2);
  __bf16* y_bf     = (__bf16*)alloc((size_t)BT * Cc * 2);

  // 1) convert activations/weights to bf16
  cvt_f32_bf16<<<2048, 256, 0, stream>>>(x, x_bf, BT * Cc);
  cvt_f32_bf16<<<2048, 256, 0, stream>>>(Wq, wqkv_bf, Hh * Dd * Cc);
  cvt_f32_bf16<<<1024, 256, 0, stream>>>(Wk, wqkv_bf + (size_t)Hh * Dd * Cc, KVh * Dd * Cc);
  cvt_f32_bf16<<<1024, 256, 0, stream>>>(Wv, wqkv_bf + (size_t)(Hh + KVh) * Dd * Cc, KVh * Dd * Cc);
  cvt_f32_bf16<<<2048, 256, 0, stream>>>(Wproj, wproj_bf, Cc * Cc);

  // 2) fused QKV projection:  qkv = x_bf @ [Wq;Wk;Wv]^T
  gemm_bf16_nt<<<dim3(NQKV / 128, BT / 128), 256, 0, stream>>>(
      x_bf, wqkv_bf, qkv, BT, NQKV, Cc);

  // 3) gate/ve add + rope + rmsnorm -> bf16 q/k/v
  qkv_post<<<BT, 128, 0, stream>>>(x, ve, cosb, sinb, Wg, qkv, q_bf, k_bf, v_bf);

  // 4) windowed causal flash attention
  flash_attn<<<dim3(Tt / 128, Hh, Bb), 256, 0, stream>>>(q_bf, k_bf, v_bf, win, y_bf);

  // 5) output projection: out = y_bf @ Wproj^T
  gemm_bf16_nt<<<dim3(Cc / 128, BT / 128), 256, 0, stream>>>(
      y_bf, wproj_bf, out, BT, Cc, Cc);
}
